// BilinearDiscriminator_77275051589764
// MI455X (gfx1250) — compile-verified, matrix operand order checked by reading, not run on hardware
//
#include <hip/hip_runtime.h>
#include <stdint.h>

// ---------------------------------------------------------------------------
// BilinearDiscriminator for MI455X (gfx1250, wave32, WMMA):
//   out = sigmoid( (x @ W^T) @ y^T )
//   x:[16384,128] f32, y:[16384,128] f32, W:[128,128] f32 -> out:[16384,16384] f32
//
// Memory-bound on the 1 GiB fp32 output (~46 us @ 23.3 TB/s). Stage-2 GEMM
// uses v_wmma_f32_16x16x32_bf16 (bf16 operands, fp32 accum) so compute sits
// under the HBM store floor. Tile staging uses the CDNA5 async global->LDS
// path (GLOBAL_LOAD_ASYNC_TO_LDS_B128, ASYNCcnt) -- bypasses the VGPR file.
// ---------------------------------------------------------------------------

typedef __attribute__((ext_vector_type(8)))  float  v8f;
typedef __attribute__((ext_vector_type(16))) __bf16 v16bf;
typedef int v4i __attribute__((vector_size(16)));   // matches builtin param type

#define AS_GLOBAL __attribute__((address_space(1)))
#define AS_LDS    __attribute__((address_space(3)))

#define N_ROWS 16384
#define M_COLS 16384
#define DIM    128

#define BT  128   // output tile edge per workgroup
#define LDT 136   // LDS row stride in bf16 elems (272 B: 16B-aligned, bank-staggered)

__device__ __forceinline__ uint16_t f32_to_bf16_rne(float f) {
    union { float f; uint32_t u; } c; c.f = f;
    uint32_t u = c.u;
    uint32_t r = u + 0x7FFFu + ((u >> 16) & 1u);   // round-to-nearest-even
    return (uint16_t)(r >> 16);
}

// ---------------- Stage 1: x_t = x @ W^T  (fp32 compute, bf16 store) --------
__global__ __launch_bounds__(128) void stage1_xWt_bf16(
    const float* __restrict__ x, const float* __restrict__ W,
    uint16_t* __restrict__ xt)
{
    __shared__ float Xs[16 * DIM];      // 16 rows of x
    __shared__ float Ws[DIM * 132];     // full W, stride 132 -> conflict-free column reads

    const int tid = threadIdx.x;
    const int rowBase = blockIdx.x * 16;

    for (int j = 0; j < DIM; ++j) {                 // 16384 floats of W
        int idx = tid + j * 128;
        int k = idx >> 7, d = idx & 127;
        Ws[k * 132 + d] = W[idx];
    }
    #pragma unroll
    for (int j = 0; j < 16; ++j) {                  // 2048 floats of x
        int idx = tid + j * 128;
        int r = idx >> 7, d = idx & 127;
        Xs[r * DIM + d] = x[(size_t)(rowBase + r) * DIM + d];
    }
    __syncthreads();

    const int k = tid;                              // output column 0..127
    float acc[16];
    #pragma unroll
    for (int r = 0; r < 16; ++r) acc[r] = 0.0f;

    for (int d = 0; d < DIM; ++d) {
        float w = Ws[k * 132 + d];
        #pragma unroll
        for (int r = 0; r < 16; ++r)                // Xs read is an LDS broadcast
            acc[r] = fmaf(Xs[r * DIM + d], w, acc[r]);
    }
    #pragma unroll
    for (int r = 0; r < 16; ++r)
        xt[(size_t)(rowBase + r) * DIM + k] = f32_to_bf16_rne(acc[r]);
}

// ---------------- Stage 1b: y -> bf16 ---------------------------------------
__global__ __launch_bounds__(256) void convert_bf16(
    const float* __restrict__ src, uint16_t* __restrict__ dst, int n)
{
    int i = blockIdx.x * 256 + threadIdx.x;
    if (i < n) dst[i] = f32_to_bf16_rne(src[i]);
}

// ---------------- async global->LDS helpers ---------------------------------
#if defined(__has_builtin)
#  if __has_builtin(__builtin_amdgcn_global_load_async_to_lds_b128)
#    define HAVE_ASYNC_LDS 1
#  endif
#endif

__device__ __forceinline__ void wait_async_zero() {
#if defined(__has_builtin)
#  if __has_builtin(__builtin_amdgcn_s_wait_asynccnt)
    __builtin_amdgcn_s_wait_asynccnt(0);
    return;
#  endif
#endif
    asm volatile("s_wait_asynccnt 0x0" ::: "memory");
}

// ---------------- Stage 2: sigmoid(x_t @ y^T) with bf16 WMMA ----------------
__global__ __launch_bounds__(256) void bilinear_gemm_sigmoid(
    const uint16_t* __restrict__ xt,   // [N,128] bf16 bits (row-major, K-major)
    const uint16_t* __restrict__ yb,   // [M,128] bf16 bits (row-major, K-major)
    float* __restrict__ out)           // [N,M]
{
    __shared__ uint16_t As[BT * LDT];  // 128 x_t rows  (34 KB)
    __shared__ uint16_t Bs[BT * LDT];  // 128 y   rows  (34 KB)

    const int tid     = threadIdx.x;
    const int colBase = blockIdx.x * BT;   // along M (y rows)
    const int rowBase = blockIdx.y * BT;   // along N (x_t rows)

    // Cooperative global->LDS staging: 2 x (128 rows * 256 B) as b128 chunks.
#if defined(HAVE_ASYNC_LDS)
    #pragma unroll
    for (int j = 0; j < 8; ++j) {
        int c   = tid + j * 256;           // chunk id 0..2047
        int row = c >> 4;                  // 16 x 16B chunks per 256 B row
        int q   = c & 15;
        const uint16_t* ga = xt + ((size_t)(rowBase + row) * DIM + q * 8);
        const uint16_t* gb = yb + ((size_t)(colBase + row) * DIM + q * 8);
        __builtin_amdgcn_global_load_async_to_lds_b128(
            (AS_GLOBAL v4i*)ga,
            (AS_LDS v4i*)&As[row * LDT + q * 8],
            0, 0);
        __builtin_amdgcn_global_load_async_to_lds_b128(
            (AS_GLOBAL v4i*)gb,
            (AS_LDS v4i*)&Bs[row * LDT + q * 8],
            0, 0);
    }
    wait_async_zero();
    __syncthreads();
#else
    #pragma unroll
    for (int j = 0; j < 8; ++j) {
        int c   = tid + j * 256;
        int row = c >> 4;
        int q   = c & 15;
        uint4 av = *(const uint4*)(xt + ((size_t)(rowBase + row) * DIM + q * 8));
        *(uint4*)(&As[row * LDT + q * 8]) = av;
        uint4 bv = *(const uint4*)(yb + ((size_t)(colBase + row) * DIM + q * 8));
        *(uint4*)(&Bs[row * LDT + q * 8]) = bv;
    }
    __syncthreads();
#endif

    // 8 waves: wave -> 32-row x 64-col sub-tile = 2x4 WMMA tiles of 16x16.
    const int wave = tid >> 5;
    const int lane = tid & 31;
    const int half = lane >> 4;            // K-half selector (ISA 16-bit A layout)
    const int r16  = lane & 15;            // row (A) / col-as-y-row (B) within tile
    const int rB   = (wave & 3) * 32;
    const int cB   = (wave >> 2) * 64;

    v8f acc[2][4];
    #pragma unroll
    for (int i = 0; i < 2; ++i)
        #pragma unroll
        for (int j = 0; j < 4; ++j)
            acc[i][j] = (v8f){0.f, 0.f, 0.f, 0.f, 0.f, 0.f, 0.f, 0.f};

    union Frag { uint4 q[2]; v16bf v; };

    #pragma unroll
    for (int kk = 0; kk < DIM; kk += 32) {
        const int k0 = kk + half * 8;      // per-lane K chunk base
        Frag a[2], b[4];
        #pragma unroll
        for (int i = 0; i < 2; ++i) {
            const uint16_t* p = &As[(rB + i * 16 + r16) * LDT + k0];
            a[i].q[0] = *(const uint4*)(p);        // K = k0 .. k0+7
            a[i].q[1] = *(const uint4*)(p + 16);   // K = k0+16 .. k0+23
        }
        #pragma unroll
        for (int j = 0; j < 4; ++j) {
            const uint16_t* p = &Bs[(cB + j * 16 + r16) * LDT + k0];
            b[j].q[0] = *(const uint4*)(p);
            b[j].q[1] = *(const uint4*)(p + 16);
        }
        #pragma unroll
        for (int i = 0; i < 2; ++i)
            #pragma unroll
            for (int j = 0; j < 4; ++j)
                acc[i][j] = __builtin_amdgcn_wmma_f32_16x16x32_bf16(
                    /*neg_a=*/false, a[i].v, /*neg_b=*/false, b[j].v,
                    /*c_mod=*/(short)0, acc[i][j],
                    /*reuse_a=*/false, /*reuse_b=*/false);
    }

    // Fused sigmoid + store. C/D layout: vgpr r -> row r + 8*half, col = r16.
    #pragma unroll
    for (int i = 0; i < 2; ++i) {
        #pragma unroll
        for (int j = 0; j < 4; ++j) {
            #pragma unroll
            for (int r = 0; r < 8; ++r) {
                int row = rowBase + rB + i * 16 + half * 8 + r;
                int col = colBase + cB + j * 16 + r16;
                float s = acc[i][j][r];
                out[(size_t)row * M_COLS + col] = 1.0f / (1.0f + __expf(-s));
            }
        }
    }
}

// ---------------------------------------------------------------------------
extern "C" void kernel_launch(void* const* d_in, const int* in_sizes, int n_in,
                              void* d_out, int out_size, void* d_ws, size_t ws_size,
                              hipStream_t stream) {
    (void)in_sizes; (void)n_in; (void)out_size; (void)ws_size;
    const float* x = (const float*)d_in[0];
    const float* y = (const float*)d_in[1];
    const float* W = (const float*)d_in[2];
    float* out = (float*)d_out;

    // workspace: [0,4MB) x_t bf16, [4MB,8MB) y bf16
    uint16_t* xt = (uint16_t*)d_ws;
    uint16_t* yb = xt + (size_t)N_ROWS * DIM;

    stage1_xWt_bf16<<<N_ROWS / 16, 128, 0, stream>>>(x, W, xt);
    convert_bf16<<<(M_COLS * DIM) / 256, 256, 0, stream>>>(y, yb, M_COLS * DIM);
    bilinear_gemm_sigmoid<<<dim3(M_COLS / BT, N_ROWS / BT), 256, 0, stream>>>(xt, yb, out);
}
